// ZoneoutLSTMEncoder_35631048687860
// MI455X (gfx1250) — compile-verified, matrix operand order checked by reading, not run on hardware
//
#include <hip/hip_runtime.h>
#include <hip/hip_bf16.h>
#include <cstdint>
#include <cstddef>

// ---------------------------------------------------------------------------
// Bidirectional zoneout-LSTM for MI455X (gfx1250), wave32 + WMMA bf16
// + async global->LDS copies (ASYNCcnt) on the recurrent critical path.
// ---------------------------------------------------------------------------

typedef __bf16 bf16_t;
typedef __attribute__((ext_vector_type(16))) __bf16 v16bf;
typedef __attribute__((ext_vector_type(8)))  float  v8f;

constexpr int T_STEPS = 1024;
constexpr int BATCH   = 32;
constexpr int IDIM    = 512;
constexpr int HDIM    = 512;
constexpr int GDIM    = 4 * HDIM;       // 2048
constexpr int NWG     = 16;             // workgroups per direction
constexpr int WPAD    = 520;            // padded LDS row (bf16 elems) = 1040B

__device__ __forceinline__ v16bf load16(const void* p0, const void* p1) {
  union { v16bf v; int4 q[2]; } u;
  u.q[0] = *(const int4*)p0;
  u.q[1] = *(const int4*)p1;
  return u.v;
}

__device__ __forceinline__ v8f wmma_bf16(v16bf a, v16bf b, v8f c) {
  // 8-arg form: (neg_a, A, neg_b, B, c_mod, C, reuse_a, reuse_b)
  return __builtin_amdgcn_wmma_f32_16x16x32_bf16(false, a, false, b,
                                                 (short)0, c, false, false);
}

__device__ __forceinline__ float sigmoidf(float x) {
  return 1.0f / (1.0f + __expf(-x));
}

// LDS byte offset of a generic pointer known to live in LDS (AS3 ptrs are 32b)
typedef __attribute__((address_space(3))) void lds_void;
__device__ __forceinline__ uint32_t lds_addr32(void* p) {
  return (uint32_t)(uintptr_t)(lds_void*)p;
}

// Async memory -> LDS copy, 16B per lane, tracked by ASYNCcnt (CDNA5 TDM-lite)
__device__ __forceinline__ void async_g2l_b128(uint32_t lds_off, const void* gaddr) {
  asm volatile("global_load_async_to_lds_b128 %0, %1, off"
               :: "v"(lds_off), "v"(gaddr) : "memory");
}
__device__ __forceinline__ void wait_asynccnt0() {
  asm volatile("s_wait_asynccnt 0x0" ::: "memory");
}

// ---------------------------------------------------------------------------
// Utility kernels
// ---------------------------------------------------------------------------
__global__ void k_init(uint32_t* bar, bf16_t* hping) {
  int i = blockIdx.x * blockDim.x + threadIdx.x;
  int n = gridDim.x * blockDim.x;
  for (int k = i; k < 2 * T_STEPS; k += n) bar[k] = 0u;
  for (int k = i; k < 4 * BATCH * HDIM; k += n) hping[k] = (bf16_t)0.0f;
}

__global__ void k_cvt4(const float4* __restrict__ s, ushort4* __restrict__ d, int n4) {
  union BU { bf16_t h; unsigned short u; };
  int i = blockIdx.x * blockDim.x + threadIdx.x;
  int stride = gridDim.x * blockDim.x;
  for (int k = i; k < n4; k += stride) {
    float4 v = s[k];
    BU a, b, c, e;
    a.h = (bf16_t)v.x; b.h = (bf16_t)v.y; c.h = (bf16_t)v.z; e.h = (bf16_t)v.w;
    ushort4 o; o.x = a.u; o.y = b.u; o.z = c.u; o.w = e.u;
    d[k] = o;
  }
}

__global__ void k_bias(const float* __restrict__ bi, const float* __restrict__ bh,
                       float* __restrict__ bs) {
  int i = blockIdx.x * blockDim.x + threadIdx.x;
  if (i < GDIM) bs[i] = bi[i] + bh[i];
}

__global__ void k_add_out(const float* __restrict__ hf, const float* __restrict__ hb,
                          float* __restrict__ out) {
  int i = blockIdx.x * blockDim.x + threadIdx.x;  // 16384*256 = exactly n/4
  float4 a = ((const float4*)hf)[i];
  float4 b = ((const float4*)hb)[i];
  float4 r;
  r.x = a.x + b.x; r.y = a.y + b.y; r.z = a.z + b.z; r.w = a.w + b.w;
  ((float4*)out)[i] = r;
}

// ---------------------------------------------------------------------------
// Phase 1: xw[T*B, 2048] = X[T*B,512](bf16) @ W_ih^T(bf16) + (b_ih+b_hh)
// Block = 128 threads (4 waves). Block tile 64(M) x 64(N); wave tile 64x16.
// ---------------------------------------------------------------------------
__global__ void k_gemm_xw(const bf16_t* __restrict__ X,    // [32768,512]
                          const bf16_t* __restrict__ W,    // [2048,512]
                          const float*  __restrict__ bsum, // [2048]
                          float* __restrict__ xw) {        // [32768,2048]
  const int lane  = threadIdx.x & 31;
  const int wv    = threadIdx.x >> 5;           // 0..3
  const int m0    = blockIdx.x * 64;
  const int n0    = blockIdx.y * 64 + wv * 16;
  const int l15   = lane & 15;
  const int khalf = (lane >> 4) << 3;           // 0 or 8 (elements)

  v8f acc[4] = {};                              // 4 M-subtiles x 1 N-tile

  const bf16_t* arow = X + (size_t)(m0 + l15) * IDIM + khalf;
  const bf16_t* brow = W + (size_t)(n0 + l15) * IDIM + khalf;

  for (int kk = 0; kk < IDIM; kk += 32) {
    if (kk + 32 < IDIM) __builtin_prefetch(brow + kk + 32, 0, 1);
    v16bf bop = load16(brow + kk, brow + kk + 16);
#pragma unroll
    for (int s = 0; s < 4; ++s) {
      const bf16_t* pa = arow + (size_t)s * 16 * IDIM + kk;
      v16bf aop = load16(pa, pa + 16);
      acc[s] = wmma_bf16(aop, bop, acc[s]);
    }
  }

  const int col = n0 + l15;
  const float bb = bsum[col];
#pragma unroll
  for (int s = 0; s < 4; ++s) {
    const int mbase = m0 + s * 16 + ((lane < 16) ? 0 : 8);
#pragma unroll
    for (int r = 0; r < 8; ++r)
      xw[(size_t)(mbase + r) * GDIM + col] = acc[s][r] + bb;
  }
}

// ---------------------------------------------------------------------------
// Phase 2: persistent recurrent scan. 32 blocks: dir = blk>>4, slab w = blk&15.
// Each block owns 32 hidden units -> 128 W_hh rows resident in LDS (bf16).
// Per step: gates[32 x 128] = xw_tile + h @ Wslab^T (WMMA), then gate math,
// zoneout, publish bf16 h slice, per-step atomic grid barrier (per direction).
// h broadcast arrives via GLOBAL_LOAD_ASYNC_TO_LDS (ASYNCcnt), overlapped with
// the xw accumulator-init loads; next step's xw tile is prefetched.
// ---------------------------------------------------------------------------
__global__ void k_scan(const float*  __restrict__ xw,     // [T*B, 2048] (+bias)
                       const bf16_t* __restrict__ Whh,    // [2048,512] bf16
                       bf16_t* __restrict__ hping,        // [2 dir][2][32*512]
                       float*  __restrict__ hs,           // [2 dir][T*B*512]
                       uint32_t* __restrict__ bar) {      // [2 dir][T]
  extern __shared__ char smem[];
  bf16_t* Wsl = (bf16_t*)smem;                    // [128][WPAD]
  bf16_t* hbf = Wsl + 128 * WPAD;                 // [32][WPAD]
  float*  gb  = (float*)(hbf + 32 * WPAD);        // [32][128]
  float*  cst = gb + 32 * 128;                    // [32][32]
  float*  hst = cst + 32 * 32;                    // [32][32]

  const int tid  = threadIdx.x;                   // 0..127
  const int lane = tid & 31;
  const int wv   = tid >> 5;                      // 0..3
  const int dir  = blockIdx.x >> 4;
  const int w    = blockIdx.x & 15;
  const int l15  = lane & 15;
  const int khalf = (lane >> 4) << 3;

  // --- W_hh slab, reordered (local row n -> global row (n>>5)*512 + w*32 + (n&31)),
  //     copied straight into LDS with async b128 transfers.
  for (int idx = tid; idx < 128 * 64; idx += 128) {   // 16B chunks
    const int n  = idx >> 6;
    const int kc = (idx & 63) << 3;
    const int g  = (n >> 5) * HDIM + w * 32 + (n & 31);
    async_g2l_b128(lds_addr32(Wsl + n * WPAD + kc), Whh + (size_t)g * HDIM + kc);
  }
  for (int i = tid; i < 32 * 32; i += 128) { cst[i] = 0.0f; hst[i] = 0.0f; }
  wait_asynccnt0();
  __syncthreads();

  const int b_el  = tid >> 2;                     // elementwise: batch row
  const int j0_el = (tid & 3) * 8;                // 8 hidden units / thread

  for (int t = 0; t < T_STEPS; ++t) {
    const int tg = dir ? (T_STEPS - 1 - t) : t;

    // 1. async broadcast h(t) bf16 -> LDS (overlaps with xw gather below)
    const bf16_t* hsrc = hping + (size_t)(dir * 2 + (t & 1)) * (BATCH * HDIM);
    for (int idx = tid; idx < 32 * 64; idx += 128) {
      const int b  = idx >> 6;
      const int kc = (idx & 63) << 3;
      async_g2l_b128(lds_addr32(hbf + b * WPAD + kc), hsrc + b * HDIM + kc);
    }

    // 2. accumulator init: acc = xw_tile (f32, includes b_ih + b_hh)
    const float* xwt = xw + (size_t)tg * BATCH * GDIM;
    v8f acc[2][2];
#pragma unroll
    for (int mt = 0; mt < 2; ++mt)
#pragma unroll
      for (int nt = 0; nt < 2; ++nt) {
        const int nloc = wv * 32 + nt * 16 + l15;
        const int gcol = (nloc >> 5) * HDIM + w * 32 + (nloc & 31);
        const int mb   = mt * 16 + ((lane < 16) ? 0 : 8);
        v8f c;
#pragma unroll
        for (int r = 0; r < 8; ++r)
          c[r] = xwt[(size_t)(mb + r) * GDIM + gcol];
        acc[mt][nt] = c;
      }

    // prefetch next step's xw tile (128 cacheline starts across the block)
    if (t + 1 < T_STEPS) {
      const int tgn = dir ? (T_STEPS - 2 - t) : (t + 1);
      const float* xwn = xw + (size_t)tgn * BATCH * GDIM;
      __builtin_prefetch(xwn + (size_t)(tid >> 2) * GDIM + (tid & 3) * HDIM + w * 32, 0, 1);
    }

    wait_asynccnt0();
    __syncthreads();

    // 3. WMMA: acc += h @ Wslab^T
    for (int kk = 0; kk < HDIM; kk += 32) {
      const bf16_t* pa0 = hbf + (0  + l15) * WPAD + khalf + kk;
      const bf16_t* pa1 = hbf + (16 + l15) * WPAD + khalf + kk;
      const bf16_t* pb0 = Wsl + (wv * 32 +      l15) * WPAD + khalf + kk;
      const bf16_t* pb1 = Wsl + (wv * 32 + 16 + l15) * WPAD + khalf + kk;
      v16bf a0 = load16(pa0, pa0 + 16);
      v16bf a1 = load16(pa1, pa1 + 16);
      v16bf b0 = load16(pb0, pb0 + 16);
      v16bf b1 = load16(pb1, pb1 + 16);
      acc[0][0] = wmma_bf16(a0, b0, acc[0][0]);
      acc[0][1] = wmma_bf16(a0, b1, acc[0][1]);
      acc[1][0] = wmma_bf16(a1, b0, acc[1][0]);
      acc[1][1] = wmma_bf16(a1, b1, acc[1][1]);
    }

    // 4. spill gate pre-activations to LDS
#pragma unroll
    for (int mt = 0; mt < 2; ++mt)
#pragma unroll
      for (int nt = 0; nt < 2; ++nt) {
        const int nloc = wv * 32 + nt * 16 + l15;
        const int mb   = mt * 16 + ((lane < 16) ? 0 : 8);
#pragma unroll
        for (int r = 0; r < 8; ++r)
          gb[(mb + r) * 128 + nloc] = acc[mt][nt][r];
      }
    __syncthreads();

    // 5. gate math + zoneout for 8 (b, j) units per thread
    bf16_t* hdst = hping + (size_t)(dir * 2 + ((t + 1) & 1)) * (BATCH * HDIM);
    float*  hsd  = hs + (size_t)dir * ((size_t)T_STEPS * BATCH * HDIM)
                      + ((size_t)tg * BATCH + b_el) * HDIM + w * 32;
#pragma unroll
    for (int jj = 0; jj < 8; ++jj) {
      const int j = j0_el + jj;
      float iv = sigmoidf(gb[b_el * 128 +       j]);
      float fv = sigmoidf(gb[b_el * 128 +  32 + j]);
      float gv = tanhf   (gb[b_el * 128 +  64 + j]);
      float ov = sigmoidf(gb[b_el * 128 +  96 + j]);
      const float co = cst[b_el * 32 + j];
      const float hp = hst[b_el * 32 + j];
      const float cn = fv * co + iv * gv;
      const float hn = ov * tanhf(cn);
      const float c2 = 0.9f * cn + 0.1f * co;   // zoneout cell
      const float h2 = 0.9f * hn + 0.1f * hp;   // zoneout hidden
      cst[b_el * 32 + j] = c2;
      hst[b_el * 32 + j] = h2;
      hdst[b_el * HDIM + w * 32 + j] = (bf16_t)h2;
      hsd[j] = h2;
    }

    // 6. per-direction grid barrier (fresh counter per step)
    __threadfence();
    __syncthreads();
    if (tid == 0) {
      uint32_t* c = bar + dir * T_STEPS + t;
      __hip_atomic_fetch_add(c, 1u, __ATOMIC_ACQ_REL, __HIP_MEMORY_SCOPE_AGENT);
      while (__hip_atomic_load(c, __ATOMIC_ACQUIRE, __HIP_MEMORY_SCOPE_AGENT) < NWG)
        __builtin_amdgcn_s_sleep(1);
    }
    __syncthreads();
  }
}

// ---------------------------------------------------------------------------
// Host-side orchestration
// ---------------------------------------------------------------------------
extern "C" void kernel_launch(void* const* d_in, const int* in_sizes, int n_in,
                              void* d_out, int out_size, void* d_ws, size_t ws_size,
                              hipStream_t stream) {
  const float* inputs = (const float*)d_in[0];   // [1024,32,512]
  const float* W_ih   = (const float*)d_in[1];   // [2048,512]
  const float* W_hh   = (const float*)d_in[2];   // [2048,512]
  const float* b_ih   = (const float*)d_in[3];   // [2048]
  const float* b_hh   = (const float*)d_in[4];   // [2048]
  float* out = (float*)d_out;                    // [1024,32,512]

  char* ws = (char*)d_ws;
  const size_t TB = (size_t)T_STEPS * BATCH;               // 32768
  size_t off = 0;
  float*    xw     = (float*)(ws + off);   off += TB * GDIM * sizeof(float);      // 256 MB
  float*    hs     = (float*)(ws + off);   off += 2 * TB * HDIM * sizeof(float);  // 128 MB
  bf16_t*   X16    = (bf16_t*)(ws + off);  off += TB * IDIM * sizeof(bf16_t);     // 32 MB
  bf16_t*   Wih16  = (bf16_t*)(ws + off);  off += (size_t)GDIM * IDIM * sizeof(bf16_t);
  bf16_t*   Whh16  = (bf16_t*)(ws + off);  off += (size_t)GDIM * HDIM * sizeof(bf16_t);
  float*    bsum   = (float*)(ws + off);   off += GDIM * sizeof(float);
  bf16_t*   hping  = (bf16_t*)(ws + off);  off += 4 * BATCH * HDIM * sizeof(bf16_t);
  uint32_t* bar    = (uint32_t*)(ws + off);

  // zero barriers + h ping-pong every launch (deterministic for graph replay)
  k_init<<<64, 256, 0, stream>>>(bar, hping);

  // f32 -> bf16 conversions (vectorized: all sizes divisible by 4)
  k_cvt4<<<4096, 256, 0, stream>>>((const float4*)inputs, (ushort4*)X16,
                                   (int)(TB * IDIM / 4));
  k_cvt4<<<1024, 256, 0, stream>>>((const float4*)W_ih, (ushort4*)Wih16,
                                   GDIM * IDIM / 4);
  k_cvt4<<<1024, 256, 0, stream>>>((const float4*)W_hh, (ushort4*)Whh16,
                                   GDIM * HDIM / 4);
  k_bias<<<8, 256, 0, stream>>>(b_ih, b_hh, bsum);

  // phase 1: input projection GEMM
  k_gemm_xw<<<dim3(TB / 64, GDIM / 64), 128, 0, stream>>>(X16, Wih16, bsum, xw);

  // phase 2: persistent bidirectional scan (dyn LDS ~187 KB / WG)
  const size_t smem = (size_t)(128 * WPAD + 32 * WPAD) * sizeof(bf16_t)
                    + (size_t)(32 * 128 + 32 * 32 + 32 * 32) * sizeof(float);
  k_scan<<<2 * NWG, 128, smem, stream>>>(xw, Whh16, hping, hs, bar);

  // out = hs_fwd + hs_bwd
  k_add_out<<<16384, 256, 0, stream>>>(hs, hs + TB * HDIM, out);
}
